// FingerNet_70042326663193
// MI455X (gfx1250) — matrix operand order repaired and checked by reading, not compiled, not run on hardware
//
#include <hip/hip_runtime.h>

typedef __attribute__((ext_vector_type(16))) __bf16 v16bf;
typedef __attribute__((ext_vector_type(8)))  float  v8f;
typedef __attribute__((ext_vector_type(4)))  unsigned int u32x4;

#define ROWS 64
#define LDW  2056            // LDS activation row stride (elements): 1028 dwords = 4 mod 64 -> conflict-free ds_load_b128
#define THREADS 256

#define WF_ELEMS   (4u * 2u * 512u * 512u)   // finger hidden weights, bf16 fragment-packed
#define WT0_ELEMS  (512u * 2048u)            // trunk first layer
#define WMID_ELEMS (7u * 512u * 512u)        // trunk mid layers
#define TOTAL_W    (WF_ELEMS + WT0_ELEMS + WMID_ELEMS)

__device__ __forceinline__ unsigned short f2bf(float v) {
  unsigned int u = __builtin_bit_cast(unsigned int, v);
  u += 0x7FFFu + ((u >> 16) & 1u);            // round-to-nearest-even
  return (unsigned short)(u >> 16);
}
__device__ __forceinline__ float bf2f(unsigned short s) {
  unsigned int u = ((unsigned int)s) << 16;
  return __builtin_bit_cast(float, u);
}

union FragU { u32x4 u[2]; v16bf v; };

// Map flat index within one fragment-packed matrix to (n, k) source coords.
// Fragment = (ntile, ktile); inside: 32 lanes x 16 contiguous bf16.
// B layout used here (16-bit, 32x16, contiguous k): lanes 0-15 -> col N=lane, K=0..15;
// lanes 16-31 -> col N=lane-16, K=16..31.
__device__ __forceinline__ void frag_coords(unsigned m, unsigned nkt,
                                            unsigned& n, unsigned& k) {
  unsigned frag   = m >> 9;          // 512 elements per fragment
  unsigned within = m & 511u;
  unsigned lane   = within >> 4;
  unsigned e      = within & 15u;
  unsigned nt = frag / nkt;
  unsigned kt = frag % nkt;
  n = nt * 16u + (lane & 15u);
  k = kt * 32u + (lane >> 4) * 16u + e;
}

// ---- Weight pre-pack: f32 row-major -> bf16 fragment-major, zero padded ----
__global__ void fingernet_prep(const float* __restrict__ fwh,
                               const float* __restrict__ tw0,
                               const float* __restrict__ tw_mid,
                               unsigned short* __restrict__ W) {
  for (unsigned idx = blockIdx.x * blockDim.x + threadIdx.x; idx < TOTAL_W;
       idx += gridDim.x * blockDim.x) {
    float v = 0.0f;
    if (idx < WF_ELEMS) {
      // fwh shape (4,2,500,500): weight[il][k][g=n]
      unsigned il = idx >> 18;                 // i*2 + l
      unsigned m  = idx & 0x3FFFFu;
      unsigned n, k; frag_coords(m, 16u, n, k);
      if (n < 500u && k < 500u)
        v = fwh[((size_t)il * 500u + k) * 500u + n];
    } else if (idx < WF_ELEMS + WT0_ELEMS) {
      // tw0 shape (2000,500): K space is 4 padded slices of 512
      unsigned m = idx - WF_ELEMS;
      unsigned n, k; frag_coords(m, 64u, n, k); // k in 0..2047
      unsigned fi = k >> 9, f = k & 511u;
      if (n < 500u && f < 500u)
        v = tw0[((size_t)fi * 500u + f) * 500u + n];
    } else {
      // tw_mid shape (7,500,500)
      unsigned m2 = idx - WF_ELEMS - WT0_ELEMS;
      unsigned l  = m2 >> 18;
      unsigned m  = m2 & 0x3FFFFu;
      unsigned n, k; frag_coords(m, 16u, n, k);
      if (n < 500u && k < 500u)
        v = tw_mid[((size_t)l * 500u + k) * 500u + n];
    }
    W[idx] = f2bf(v);
  }
}

// ---- One fused in-place GEMM layer over a 64-row LDS tile ----
// Wave w owns ntiles 4w..4w+3 across ALL 4 mtiles (16 accumulators), so every
// B fragment is loaded from L2 exactly once per workgroup (64 FLOP/L2-byte).
// In-place: full-layer accumulation in registers, barrier, write-back, barrier.
__device__ __forceinline__ void gemm_layer_ip(
    const unsigned short* __restrict__ W, int nkt,
    const float* __restrict__ bias,
    unsigned short* __restrict__ actL) {   // LDS in/out base (incl. col offset)
  const int tid  = threadIdx.x;
  const int lane = tid & 31;
  const int wave = tid >> 5;        // 8 waves (wave32)
  const int l15  = lane & 15;
  const int lh   = lane >> 4;
  const int nt0  = wave * 4;        // 8 waves x 4 ntiles = 32 ntiles (N=512)

  v8f acc[4][4];                    // [mtile][ntile]
#pragma unroll
  for (int mt = 0; mt < 4; ++mt)
#pragma unroll
    for (int t = 0; t < 4; ++t) { v8f z = {}; acc[mt][t] = z; }

  // A fragment: lane holds row M=lane%16;
  // elems 0..15 = K kb..kb+15, kb=(lane/16)*16 (matches packed B order).
  const unsigned short* aBase = actL + (size_t)l15 * LDW + lh * 16;

  for (int kt = 0; kt < nkt; ++kt) {
    FragU fa[4];
#pragma unroll
    for (int mt = 0; mt < 4; ++mt) {
      const unsigned short* ap = aBase + (size_t)(mt * 16) * LDW + (kt << 5);
      fa[mt].u[0] = *(const u32x4*)(ap);
      fa[mt].u[1] = *(const u32x4*)(ap + 8);
    }
#pragma unroll
    for (int t = 0; t < 4; ++t) {
      const unsigned short* bp =
          W + ((((size_t)(nt0 + t) * nkt + kt) << 9) + (lane << 4));
      FragU fb;
      fb.u[0] = *(const u32x4*)(bp);
      fb.u[1] = *(const u32x4*)(bp + 8);
#pragma unroll
      for (int mt = 0; mt < 4; ++mt)
        acc[mt][t] = __builtin_amdgcn_wmma_f32_16x16x32_bf16(
            false, fa[mt].v, false, fb.v, (short)0, acc[mt][t], false, false);
    }
  }

  __syncthreads();   // all waves done reading the activation tile

  // epilogue: bias + relu + bf16 write-back (padded cols n>=500 -> relu(0)=0)
#pragma unroll
  for (int t = 0; t < 4; ++t) {
    const int ncol = (nt0 + t) * 16 + l15;
    const float bv = (ncol < 500) ? bias[ncol] : 0.0f;
#pragma unroll
    for (int mt = 0; mt < 4; ++mt) {
      const int rbase = mt * 16 + lh * 8;  // C layout: VGPR r -> M = r + (lane/16)*8
#pragma unroll
      for (int r = 0; r < 8; ++r) {
        float vv = acc[mt][t][r] + bv;
        vv = vv > 0.0f ? vv : 0.0f;
        actL[(size_t)(rbase + r) * LDW + ncol] = f2bf(vv);
      }
    }
  }
  __syncthreads();   // writes visible before next layer reads
}

// ---- Fully fused network: each workgroup carries 64 rows through all layers ----
__global__ __launch_bounds__(THREADS, 1)
void fingernet_main(const float* __restrict__ x,
                    const float* __restrict__ lb, const float* __restrict__ ub,
                    const float* __restrict__ fw1, const float* __restrict__ fb1,
                    const float* __restrict__ fbh,
                    const float* __restrict__ tb0,
                    const float* __restrict__ tb_mid,
                    const float* __restrict__ tw_out,
                    const float* __restrict__ tb_out,
                    const unsigned short* __restrict__ WF,
                    const unsigned short* __restrict__ WT0,
                    const unsigned short* __restrict__ WMID,
                    float* __restrict__ out) {
  extern __shared__ char smem[];
  unsigned short* buf = (unsigned short*)smem;         // [64][2056] bf16, in-place
  float* xn   = (float*)(buf + ROWS * LDW);            // [64][4]
  float* oacc = xn + ROWS * 4;                         // [64]

  const int tid = threadIdx.x;
  const int r0  = blockIdx.x * ROWS;

  {  // normalize inputs (ROWS*4 == THREADS)
    int b = tid >> 2, i = tid & 3;
    float xv = x[(size_t)(r0 + b) * 4 + i];
    xn[tid] = 2.0f * (xv - lb[i]) / (ub[i] - lb[i]) - 1.0f;
  }
  if (tid < ROWS) oacc[tid] = 0.0f;
  __syncthreads();

  // Finger layer 1: h[b, i*512+f] = relu(xn[b,i]*fw1[i,f] + fb1[i,f]); pad f>=500 -> 0
  for (int idx = tid; idx < ROWS * 2048; idx += THREADS) {
    int b = idx >> 11, c = idx & 2047, i = c >> 9, f = c & 511;
    float v = 0.0f;
    if (f < 500) {
      float z = xn[b * 4 + i] * fw1[i * 500 + f] + fb1[i * 500 + f];
      v = z > 0.0f ? z : 0.0f;
    }
    buf[(size_t)b * LDW + c] = f2bf(v);
  }
  __syncthreads();

  // Finger hidden layers (NFL-1 = 2): 4 grouped 64x512x512 GEMMs each, in-place per slice
  for (int l = 0; l < 2; ++l)
    for (int i = 0; i < 4; ++i)
      gemm_layer_ip(WF + ((size_t)(i * 2 + l) << 18), 16,
                    fbh + (size_t)(i * 2 + l) * 500, buf + i * 512);

  // Trunk first layer: K = 2048 (concat of 4 padded finger slices) -> cols 0..511
  gemm_layer_ip(WT0, 64, tb0, buf);

  // Trunk mid layers (NL-1 = 7): 64x512x512, in-place
  for (int l = 0; l < 7; ++l)
    gemm_layer_ip(WMID + ((size_t)l << 18), 16, tb_mid + (size_t)l * 500, buf);

  // Output layer: OUT=1 dot product per row
  {
    int b = tid >> 2, p = tid & 3;
    float s = 0.0f;
    for (int k = p; k < 500; k += 4)
      s += bf2f(buf[(size_t)b * LDW + k]) * tw_out[k];
    atomicAdd(&oacc[b], s);
  }
  __syncthreads();
  if (tid < ROWS) out[r0 + tid] = oacc[tid] + tb_out[0];  // SCALING = 1.0
}

extern "C" void kernel_launch(void* const* d_in, const int* in_sizes, int n_in,
                              void* d_out, int out_size, void* d_ws, size_t ws_size,
                              hipStream_t stream) {
  (void)in_sizes; (void)n_in; (void)out_size; (void)ws_size;
  const float* x      = (const float*)d_in[0];
  const float* lb     = (const float*)d_in[1];
  const float* ub     = (const float*)d_in[2];
  const float* fw1    = (const float*)d_in[3];
  const float* fb1    = (const float*)d_in[4];
  const float* fwh    = (const float*)d_in[5];
  const float* fbh    = (const float*)d_in[6];
  const float* tw0    = (const float*)d_in[7];
  const float* tb0    = (const float*)d_in[8];
  const float* tw_mid = (const float*)d_in[9];
  const float* tb_mid = (const float*)d_in[10];
  const float* tw_out = (const float*)d_in[11];
  const float* tb_out = (const float*)d_in[12];

  unsigned short* WF   = (unsigned short*)d_ws;       // ~9.5 MB bf16 total in ws
  unsigned short* WT0p = WF + WF_ELEMS;
  unsigned short* WMID = WT0p + WT0_ELEMS;

  fingernet_prep<<<4096, 256, 0, stream>>>(fwh, tw0, tw_mid, WF);

  const size_t shmem = (size_t)ROWS * LDW * sizeof(unsigned short)
                     + (size_t)ROWS * 4 * sizeof(float)
                     + (size_t)ROWS * sizeof(float);   // ~264 KB (<320 KB WGP LDS)
  hipFuncSetAttribute((const void*)fingernet_main,
                      hipFuncAttributeMaxDynamicSharedMemorySize, (int)shmem);

  fingernet_main<<<32768 / ROWS, THREADS, shmem, stream>>>(
      x, lb, ub, fw1, fb1, fbh, tb0, tb_mid, tw_out, tb_out,
      WF, WT0p, WMID, (float*)d_out);
}
